// QGNNGraphClassifierTFQ_8383776162481
// MI455X (gfx1250) — matrix operand order repaired and checked by reading, not compiled, not run on hardware
//
#include <hip/hip_runtime.h>
#include <math.h>

typedef __attribute__((ext_vector_type(16))) _Float16 v16h;
typedef __attribute__((ext_vector_type(8)))  _Float16 v8h;
typedef __attribute__((ext_vector_type(8)))  float    v8f;

#define PI_F 3.14159265358979323846f

// ---------------------------------------------------------------------------
// Generic 2-layer MLP: Y = act2( lrelu(X @ W1 + b1) @ W2 + b2 )
// KIN in {16, 8, 3}, hidden = 128, out = 2.  One wave per 16-row tile,
// WMMA f16 -> f32.  All global/LDS loads are unconditional (clamped indices +
// zero masks) so no per-element exec-mask branching.
// ---------------------------------------------------------------------------
template<int KIN, bool TANH, bool SCATTER>
__global__ __launch_bounds__(128)
void mlp_wmma_kernel(const float* __restrict__ X, int M,
                     const float* __restrict__ W1, const float* __restrict__ b1,
                     const float* __restrict__ W2, const float* __restrict__ b2,
                     float* __restrict__ Y, const int* __restrict__ sidx)
{
    __shared__ _Float16 Hs[4][16 * 128];   // per-wave hidden tile (f16)
    const int wave = threadIdx.x >> 5;
    const int lane = threadIdx.x & 31;
    const int tile = blockIdx.x * 4 + wave;
    const int row0 = tile * 16;
    const int m    = lane & 15;            // row inside tile (A row id)
    const int hi   = lane >> 4;            // lane-half select
    int rowg = row0 + m;
    const float rmask = (rowg < M) ? 1.0f : 0.0f;
    if (rowg >= M) rowg = M - 1;           // clamp: loads always in-bounds

    // --- A tile: 16 x 32 f16 (ISA 16-bit A layout); valid K are e<8 with
    //     k = hi*8 + e < KIN; everything else compile-time zero. ------------
    v16h A;
#pragma unroll
    for (int e = 0; e < 16; ++e) A[e] = (_Float16)0.0f;

    if constexpr (KIN == 16) {
        const float4* Xv = (const float4*)X;
        float4 a0 = Xv[(size_t)rowg * 4 + hi * 2 + 0];   // k = hi*8 .. +3
        float4 a1 = Xv[(size_t)rowg * 4 + hi * 2 + 1];   // k = hi*8+4 .. +7
        A[0] = (_Float16)(a0.x * rmask); A[1] = (_Float16)(a0.y * rmask);
        A[2] = (_Float16)(a0.z * rmask); A[3] = (_Float16)(a0.w * rmask);
        A[4] = (_Float16)(a1.x * rmask); A[5] = (_Float16)(a1.y * rmask);
        A[6] = (_Float16)(a1.z * rmask); A[7] = (_Float16)(a1.w * rmask);
    } else if constexpr (KIN == 8) {
        const float4* Xv = (const float4*)X;
        float msk = (hi == 0) ? rmask : 0.0f;            // hi half has k>=8
        float4 a0 = Xv[(size_t)rowg * 2 + 0];
        float4 a1 = Xv[(size_t)rowg * 2 + 1];
        A[0] = (_Float16)(a0.x * msk); A[1] = (_Float16)(a0.y * msk);
        A[2] = (_Float16)(a0.z * msk); A[3] = (_Float16)(a0.w * msk);
        A[4] = (_Float16)(a1.x * msk); A[5] = (_Float16)(a1.y * msk);
        A[6] = (_Float16)(a1.z * msk); A[7] = (_Float16)(a1.w * msk);
    } else {  // KIN == 3
        float msk = (hi == 0) ? rmask : 0.0f;
        float x0 = X[(size_t)rowg * 3 + 0];
        float x1 = X[(size_t)rowg * 3 + 1];
        float x2 = X[(size_t)rowg * 3 + 2];
        A[0] = (_Float16)(x0 * msk);
        A[1] = (_Float16)(x1 * msk);
        A[2] = (_Float16)(x2 * msk);
    }

    // --- hidden layer: 8 column tiles of 16 -> 8 WMMAs --------------------
    const int n = lane & 15;
    const _Float16 hzero = (_Float16)0.0f;
#pragma unroll
    for (int nt = 0; nt < 8; ++nt) {
        const int col = nt * 16 + n;
        // B: table order, k = hi*8 + s for s<8; valid only for k<KIN (KIN<=16).
        v16h B;
#pragma unroll
        for (int e = 0; e < 16; ++e) B[e] = hzero;
#pragma unroll
        for (int e = 0; e < KIN; ++e) {
            float w = W1[e * 128 + col];                 // unconditional load
            B[e & 7] = (hi == (e >> 3)) ? (_Float16)w : B[e & 7];
        }
        v8f C = {};
        C = __builtin_amdgcn_wmma_f32_16x16x32_f16(false, A, false, B,
                                                   (short)0, C, false, false);
        float bias = b1[col];
#pragma unroll
        for (int r = 0; r < 8; ++r) {
            float hv = C[r] + bias;
            hv = hv > 0.0f ? hv : 0.2f * hv;             // leaky_relu(0.2)
            Hs[wave][(hi * 8 + r) * 128 + col] = (_Float16)hv;
        }
    }
    __syncthreads();

    // --- output layer: H(16x128) @ W2(128x2), K in 4 chunks of 32 ---------
    const v8h* Hrow = (const v8h*)&Hs[wave][m * 128];
    const int ncl = n & 1;                               // clamped W2 column
    v8f C2 = {};
#pragma unroll
    for (int kc = 0; kc < 4; ++kc) {
        // A2: e<8 -> k = kc*32+hi*8+e (contiguous), e>=8 -> +16 (contiguous)
        v8h lo = Hrow[kc * 4 + hi];
        v8h hb = Hrow[kc * 4 + hi + 2];
        v16h A2;
#pragma unroll
        for (int j = 0; j < 8; ++j) { A2[j] = lo[j]; A2[8 + j] = hb[j]; }
        v16h B2;
#pragma unroll
        for (int e = 0; e < 16; ++e) {
            int k = kc * 32 + hi * 8 + (e & 7) + (e >> 3) * 16; // always < 128
            float w = W2[k * 2 + ncl];                   // unconditional load
            B2[e] = (n < 2) ? (_Float16)w : hzero;
        }
        C2 = __builtin_amdgcn_wmma_f32_16x16x32_f16(false, A2, false, B2,
                                                    (short)0, C2, false, false);
    }

    if (n < 2) {
        float bias = b2[n];
#pragma unroll
        for (int r = 0; r < 8; ++r) {
            int rg = row0 + hi * 8 + r;
            if (rg < M) {
                float v = C2[r] + bias;
                if (TANH) v = tanhf(v) * PI_F;
                if (SCATTER) {
                    int g = sidx[rg * 4 + 0]; if (g < 0) g = 0;
                    atomicAdd(&Y[g * 2 + n], v);
                } else {
                    Y[rg * 2 + n] = v;
                }
            }
        }
    }
}

// ---------------------------------------------------------------------------
// 9-qubit statevector PQC, one wave per node.
// Flat amp index idx = lane*16 + t  (lane = bits 8..4, t = bits 3..0).
// Qubit q <-> flat bit (8-q): q in 5..8 lane-local, q in 0..4 cross-lane.
// ---------------------------------------------------------------------------
__device__ __forceinline__ float lx(float v, int mask) { return __shfl_xor(v, mask, 32); }

template<int Q>
__device__ __forceinline__ bool qbit(int lane, int t) {
    constexpr int b = 8 - Q;
    if constexpr (b <= 3) return ((t >> b) & 1) != 0;
    else                  return ((lane >> (b - 4)) & 1) != 0;
}

template<int Q, int CQ>
__device__ __forceinline__ void applyU(float (&sr)[16], float (&si)[16], int lane,
    float r00, float i00, float r01, float i01,
    float r10, float i10, float r11, float i11)
{
    constexpr int B = 8 - Q;
    if constexpr (B <= 3) {                       // lane-local pairing
        constexpr int mt = 1 << B;
#pragma unroll
        for (int t = 0; t < 16; ++t) {
            if ((t & mt) != 0) continue;
            const int t1 = t | mt;
            bool act = true;
            if constexpr (CQ >= 0) act = qbit<CQ>(lane, t);
            if (act) {
                float a0r = sr[t],  a0i = si[t];
                float a1r = sr[t1], a1i = si[t1];
                sr[t]  = r00*a0r - i00*a0i + r01*a1r - i01*a1i;
                si[t]  = r00*a0i + i00*a0r + r01*a1i + i01*a1r;
                sr[t1] = r10*a0r - i10*a0i + r11*a1r - i11*a1i;
                si[t1] = r10*a0i + i10*a0r + r11*a1i + i11*a1r;
            }
        }
    } else {                                      // cross-lane pairing
        constexpr int ml = 1 << (B - 4);
        const bool hiSide = (lane & ml) != 0;
#pragma unroll
        for (int t = 0; t < 16; ++t) {
            float pr = lx(sr[t], ml);             // partner amp (unconditional)
            float pi = lx(si[t], ml);
            bool act = true;
            if constexpr (CQ >= 0) act = qbit<CQ>(lane, t);
            float ar = sr[t], ai = si[t];
            float nr, ni;
            if (!hiSide) { nr = r00*ar - i00*ai + r01*pr - i01*pi;
                           ni = r00*ai + i00*ar + r01*pi + i01*pr; }
            else         { nr = r10*pr - i10*pi + r11*ar - i11*ai;
                           ni = r10*pi + i10*pr + r11*ai + i11*ar; }
            if (act) { sr[t] = nr; si[t] = ni; }
        }
    }
}

template<int Q, int CQ = -1>
__device__ __forceinline__ void RY(float (&sr)[16], float (&si)[16], int lane, float th) {
    float c = cosf(0.5f * th), s = sinf(0.5f * th);
    applyU<Q, CQ>(sr, si, lane, c, 0.f, -s, 0.f, s, 0.f, c, 0.f);
}
template<int Q, int CQ = -1>
__device__ __forceinline__ void RX(float (&sr)[16], float (&si)[16], int lane, float th) {
    float c = cosf(0.5f * th), s = sinf(0.5f * th);
    applyU<Q, CQ>(sr, si, lane, c, 0.f, 0.f, -s, 0.f, -s, c, 0.f);
}
template<int Q, int CQ = -1>
__device__ __forceinline__ void RZ(float (&sr)[16], float (&si)[16], int lane, float th) {
    float c = cosf(0.5f * th), s = sinf(0.5f * th);
    applyU<Q, CQ>(sr, si, lane, c, -s, 0.f, 0.f, 0.f, 0.f, c, s);
}
template<int QA, int QB>
__device__ __forceinline__ void CZ(float (&sr)[16], float (&si)[16], int lane) {
#pragma unroll
    for (int t = 0; t < 16; ++t)
        if (qbit<QA>(lane, t) && qbit<QB>(lane, t)) { sr[t] = -sr[t]; si[t] = -si[t]; }
}

__global__ __launch_bounds__(256)
void pqc_kernel(const float* __restrict__ node_f, const float* __restrict__ edge_f,
                const int* __restrict__ gnodes, const int* __restrict__ gedges,
                const float* __restrict__ theta, float* __restrict__ expv, int N)
{
    const int lane = threadIdx.x & 31;
    const int node = blockIdx.x * (blockDim.x >> 5) + (threadIdx.x >> 5);
    if (node >= N) return;                        // uniform per wave

    float data[18];
#pragma unroll
    for (int j = 0; j < 3; ++j) {
        int ge = gedges[node * 3 + j];
        float msk = ge >= 0 ? 1.0f : 0.0f;
        int gc = ge < 0 ? 0 : ge;
        data[2 * j + 0] = edge_f[gc * 2 + 0] * msk;
        data[2 * j + 1] = edge_f[gc * 2 + 1] * msk;
    }
#pragma unroll
    for (int j = 0; j < 4; ++j) {
        int gv = gnodes[node * 4 + j];
        float msk = gv >= 0 ? 1.0f : 0.0f;
        int gc = gv < 0 ? 0 : gv;
        data[6 + 2 * j + 0] = node_f[gc * 2 + 0] * msk;
        data[6 + 2 * j + 1] = node_f[gc * 2 + 1] * msk;
    }
    data[14] = data[8];  data[15] = data[0];      // phis
    data[16] = data[9];  data[17] = data[1];

    float th[27];
#pragma unroll
    for (int i = 0; i < 27; ++i) th[i] = theta[i];

    float sr[16], si[16];
#pragma unroll
    for (int t = 0; t < 16; ++t) { sr[t] = 0.f; si[t] = 0.f; }
    if (lane == 0) sr[0] = 1.0f;                  // |0...0>

    // data encoding layer
    RY<0>(sr, si, lane, data[0]);  RZ<0>(sr, si, lane, data[1]);
    RY<1>(sr, si, lane, data[2]);  RZ<1>(sr, si, lane, data[3]);
    RY<2>(sr, si, lane, data[4]);  RZ<2>(sr, si, lane, data[5]);
    RY<3>(sr, si, lane, data[6]);  RZ<3>(sr, si, lane, data[7]);
    RY<4>(sr, si, lane, data[8]);  RZ<4>(sr, si, lane, data[9]);
    RY<5>(sr, si, lane, data[10]); RZ<5>(sr, si, lane, data[11]);
    RY<6>(sr, si, lane, data[12]); RZ<6>(sr, si, lane, data[13]);
    // controlled encoding
    RX<7, 4>(sr, si, lane, data[14]);
    RY<7, 0>(sr, si, lane, data[15]);
    RZ<8, 4>(sr, si, lane, data[16]);
    RY<8, 0>(sr, si, lane, data[17]);
    // variational block 0: qs = (0,4,7)
    RX<0>(sr, si, lane, th[0]);  RY<0>(sr, si, lane, th[1]);  RZ<0>(sr, si, lane, th[2]);
    RX<4>(sr, si, lane, th[3]);  RY<4>(sr, si, lane, th[4]);  RZ<4>(sr, si, lane, th[5]);
    RX<7>(sr, si, lane, th[6]);  RY<7>(sr, si, lane, th[7]);  RZ<7>(sr, si, lane, th[8]);
    CZ<0, 4>(sr, si, lane); CZ<4, 7>(sr, si, lane); CZ<7, 0>(sr, si, lane);
    // block 1: qs = (7,4,8)
    RX<7>(sr, si, lane, th[9]);  RY<7>(sr, si, lane, th[10]); RZ<7>(sr, si, lane, th[11]);
    RX<4>(sr, si, lane, th[12]); RY<4>(sr, si, lane, th[13]); RZ<4>(sr, si, lane, th[14]);
    RX<8>(sr, si, lane, th[15]); RY<8>(sr, si, lane, th[16]); RZ<8>(sr, si, lane, th[17]);
    CZ<7, 4>(sr, si, lane); CZ<4, 8>(sr, si, lane); CZ<8, 7>(sr, si, lane);
    // block 2: qs = (3,7,8)
    RX<3>(sr, si, lane, th[18]); RY<3>(sr, si, lane, th[19]); RZ<3>(sr, si, lane, th[20]);
    RX<7>(sr, si, lane, th[21]); RY<7>(sr, si, lane, th[22]); RZ<7>(sr, si, lane, th[23]);
    RX<8>(sr, si, lane, th[24]); RY<8>(sr, si, lane, th[25]); RZ<8>(sr, si, lane, th[26]);
    CZ<3, 7>(sr, si, lane); CZ<7, 8>(sr, si, lane); CZ<8, 3>(sr, si, lane);

    // exp = Re<psi| X_q3 |psi>; qubit 3 -> flat bit 5 -> lane-xor 2
    float acc = 0.0f;
#pragma unroll
    for (int t = 0; t < 16; ++t) {
        float pr = lx(sr[t], 2);
        float pi = lx(si[t], 2);
        acc += sr[t] * pr + si[t] * pi;
    }
    for (int msk = 16; msk >= 1; msk >>= 1) acc += lx(acc, msk);
    if (lane == 0) expv[node] = acc;
}

// ---------------------------------------------------------------------------
__global__ void prep_kernel(const float* __restrict__ nf, const float* __restrict__ expv,
                            float* __restrict__ updX, float* __restrict__ nodeB, int N)
{
    int i = blockIdx.x * blockDim.x + threadIdx.x;
    if (i < N) {
        float a = nf[2 * i], b = nf[2 * i + 1];
        updX[3 * i + 0] = a; updX[3 * i + 1] = b; updX[3 * i + 2] = expv[i];
        nodeB[2 * i + 0] = a; nodeB[2 * i + 1] = b;
    }
}

// Deterministic per-graph mean pooling + head MLP (one thread per graph).
__global__ void head_kernel(const float* __restrict__ nf, const int* __restrict__ batch, int N,
                            const float* __restrict__ W1, const float* __restrict__ b1,
                            const float* __restrict__ W2, const float* __restrict__ b2,
                            float* __restrict__ out, int G)
{
    int g = blockIdx.x * blockDim.x + threadIdx.x;
    if (g >= G) return;
    float s0 = 0.f, s1 = 0.f, c = 0.f;
    for (int i = 0; i < N; ++i)
        if (batch[i] == g) { s0 += nf[2 * i]; s1 += nf[2 * i + 1]; c += 1.0f; }
    float inv = c > 0.f ? 1.0f / c : 0.0f;
    float e0 = s0 * inv, e1 = s1 * inv;
    float h0 = e0 * W1[0] + e1 * W1[2] + b1[0];
    float h1 = e0 * W1[1] + e1 * W1[3] + b1[1];
    h0 = h0 > 0.f ? h0 : 0.2f * h0;
    h1 = h1 > 0.f ? h1 : 0.2f * h1;
    out[2 * g + 0] = h0 * W2[0] + h1 * W2[2] + b2[0];
    out[2 * g + 1] = h0 * W2[1] + h1 * W2[3] + b2[1];
}

// ---------------------------------------------------------------------------
extern "C" void kernel_launch(void* const* d_in, const int* in_sizes, int n_in,
                              void* d_out, int out_size, void* d_ws, size_t ws_size,
                              hipStream_t stream)
{
    const float* node_feat = (const float*)d_in[0];
    const float* edge_attr = (const float*)d_in[1];
    const float* node_W1   = (const float*)d_in[2];
    const float* node_b1   = (const float*)d_in[3];
    const float* node_W2   = (const float*)d_in[4];
    const float* node_b2   = (const float*)d_in[5];
    const float* edge_W1   = (const float*)d_in[6];
    const float* edge_b1   = (const float*)d_in[7];
    const float* edge_W2   = (const float*)d_in[8];
    const float* edge_b2   = (const float*)d_in[9];
    const float* theta     = (const float*)d_in[10];
    const float* upd_W1    = (const float*)d_in[11];
    const float* upd_b1    = (const float*)d_in[12];
    const float* upd_W2    = (const float*)d_in[13];
    const float* upd_b2    = (const float*)d_in[14];
    const float* head_W1   = (const float*)d_in[15];
    const float* head_b1   = (const float*)d_in[16];
    const float* head_W2   = (const float*)d_in[17];
    const float* head_b2   = (const float*)d_in[18];
    const int*   gnodes    = (const int*)d_in[19];
    const int*   gedges    = (const int*)d_in[20];
    const int*   batch     = (const int*)d_in[21];

    const int N = in_sizes[0] / 16;
    const int E = in_sizes[1] / 8;
    const int G = out_size / 2;
    int H = in_sizes[10] / 27; if (H < 1) H = 1;

    float* ws     = (float*)d_ws;
    float* nodeA  = ws;                      // N*2
    float* nodeB  = nodeA + (size_t)N * 2;   // N*2
    float* edge_f = nodeB + (size_t)N * 2;   // E*2
    float* expv   = edge_f + (size_t)E * 2;  // N
    float* updX   = expv + (size_t)N;        // N*3

    // node / edge feature MLPs (WMMA)
    mlp_wmma_kernel<16, true, false><<<(N + 63) / 64, 128, 0, stream>>>(
        node_feat, N, node_W1, node_b1, node_W2, node_b2, nodeA, nullptr);
    mlp_wmma_kernel<8, true, false><<<(E + 63) / 64, 128, 0, stream>>>(
        edge_attr, E, edge_W1, edge_b1, edge_W2, edge_b2, edge_f, nullptr);

    float* cur = nodeA;
    float* nxt = nodeB;
    for (int h = 0; h < H; ++h) {
        pqc_kernel<<<(N + 7) / 8, 256, 0, stream>>>(cur, edge_f, gnodes, gedges,
                                                    theta + (size_t)h * 27, expv, N);
        prep_kernel<<<(N + 255) / 256, 256, 0, stream>>>(cur, expv, updX, nxt, N);
        mlp_wmma_kernel<3, false, true><<<(N + 63) / 64, 128, 0, stream>>>(
            updX, N, upd_W1 + (size_t)h * 3 * 128, upd_b1 + (size_t)h * 128,
            upd_W2 + (size_t)h * 128 * 2, upd_b2 + (size_t)h * 2, nxt, gnodes);
        float* t = cur; cur = nxt; nxt = t;
    }

    head_kernel<<<(G + 63) / 64, 64, 0, stream>>>(cur, batch, N, head_W1, head_b1,
                                                  head_W2, head_b2, (float*)d_out, G);
}
